// TemporalSelfAttentionLayer_94489281071
// MI455X (gfx1250) — matrix operand order repaired and checked by reading, not compiled, not run on hardware
//
#include <hip/hip_runtime.h>
#include <hip/hip_bf16.h>

// ---------------------------------------------------------------------------
// Types for CDNA5 WMMA
// ---------------------------------------------------------------------------
typedef __attribute__((ext_vector_type(16))) __bf16 v16bf;
typedef __attribute__((ext_vector_type(8)))  float  v8f;

union FragU {
    v16bf v;
    unsigned int   w[8];
    unsigned short u[16];
};

__device__ __forceinline__ unsigned short f2bf(float f) {
    unsigned int u = __float_as_uint(f);
    u += 0x7FFFu + ((u >> 16) & 1u);      // round-to-nearest-even
    return (unsigned short)(u >> 16);
}

__device__ __forceinline__ v8f wmma_bf16(v16bf a, v16bf b, v8f c) {
    // D = A(16x32) * B(32x16) + C, f32 accumulate
    return __builtin_amdgcn_wmma_f32_16x16x32_bf16(false, a, false, b, (short)0, c, false, false);
}

// CDNA5 async global->LDS copy (ASYNCcnt-tracked, no VGPR round trip).
__device__ __forceinline__ void async_ld_b128(unsigned short* lds_dst, const unsigned short* gsrc) {
    unsigned lds = (unsigned)(uintptr_t)(void*)lds_dst;   // low 32 bits = LDS offset
    asm volatile("global_load_async_to_lds_b128 %0, %1, off"
                 :: "v"(lds), "v"(gsrc)
                 : "memory");
}
__device__ __forceinline__ void wait_async0() {
    asm volatile("s_wait_asynccnt 0" ::: "memory");
}

// A fragment: 16x32 bf16. lane%16 = row M; lane/16 selects K-half.
// element i: K = h*16 + i.
__device__ __forceinline__ v16bf load_a_frag(const unsigned short* row0, int stride, int lane) {
    const int m = lane & 15;
    const int h = (lane >> 4) & 1;
    const unsigned short* p = row0 + m * stride;
    FragU f;
#pragma unroll
    for (int j = 0; j < 4; ++j)
        f.w[j] = *(const unsigned int*)(p + h * 16 + 2 * j);
#pragma unroll
    for (int j = 0; j < 4; ++j)
        f.w[4 + j] = *(const unsigned int*)(p + h * 16 + 8 + 2 * j);
    return f.v;
}

// B fragment: 32x16 bf16 from an [N][K] (row=N, contiguous K) buffer.
// lane%16 = column N; element i: K = (lane/16)*16 + i.
__device__ __forceinline__ v16bf load_b_frag_nk(const unsigned short* base, int stride, int lane) {
    const int n = lane & 15;
    const int h = (lane >> 4) & 1;
    const unsigned short* p = base + n * stride + h * 16;
    FragU f;
#pragma unroll
    for (int j = 0; j < 8; ++j)
        f.w[j] = *(const unsigned int*)(p + 2 * j);
    return f.v;
}

// ---------------------------------------------------------------------------
// Problem constants
// ---------------------------------------------------------------------------
#define SEQ   1024
#define BATCH 8
#define EMB   1024
#define HEADS 16
#define HDIM  64
#define ROWS  (SEQ * BATCH)   // 8192 GEMM rows, r = s*BATCH + b

// ---------------------------------------------------------------------------
// fp32 -> bf16 weight conversion
// ---------------------------------------------------------------------------
__global__ __launch_bounds__(256) void cvt_bf16_kernel(const float* __restrict__ in,
                                                       unsigned short* __restrict__ out, int n) {
    int i = blockIdx.x * 256 + threadIdx.x;
    if (i < n) out[i] = f2bf(in[i]);
}

// ---------------------------------------------------------------------------
// QKV projection: [8192 x 1024] (fp32 x, converted in LDS) * Wqkv^T[1024 x 3072]
// scatter bias-added bf16 results to q/k/v buffers laid out [B,H,S,D].
// ---------------------------------------------------------------------------
__global__ __launch_bounds__(256) void qkv_gemm_kernel(const float* __restrict__ x,
                                                       const unsigned short* __restrict__ wb,
                                                       const float* __restrict__ bias,
                                                       unsigned short* __restrict__ qb,
                                                       unsigned short* __restrict__ kb,
                                                       unsigned short* __restrict__ vb) {
    __shared__ unsigned short lds_a[128 * 32];   // [m][k] bf16
    __shared__ unsigned short lds_b[128 * 32];   // [n][k] bf16

    const int tid  = threadIdx.x;
    const int lane = tid & 31;
    const int wave = tid >> 5;
    const int n0 = blockIdx.x * 128;
    const int r0 = blockIdx.y * 128;
    const int wm = (wave >> 1) * 32;   // 4 waves down M
    const int wn = (wave & 1) * 64;    // 2 waves across N

    v8f acc[2][4];
    const v8f vzero = {0.f, 0.f, 0.f, 0.f, 0.f, 0.f, 0.f, 0.f};
#pragma unroll
    for (int mt = 0; mt < 2; ++mt)
#pragma unroll
        for (int nt = 0; nt < 4; ++nt) acc[mt][nt] = vzero;

    for (int ks = 0; ks < EMB; ks += 32) {
        __syncthreads();
        // A tile: 128x32 fp32 -> bf16 LDS. Issue all loads first, then convert.
        {
            float4 f4[4];
#pragma unroll
            for (int j = 0; j < 4; ++j) {
                int it = tid + 256 * j;                  // 1024 float4-slots total
                int row = it >> 3, c4 = (it & 7) * 4;
                f4[j] = *(const float4*)(x + (size_t)(r0 + row) * EMB + ks + c4);
                if (ks + 32 < EMB)                        // prefetch next K-step line
                    __builtin_prefetch(x + (size_t)(r0 + row) * EMB + ks + 32 + c4, 0, 3);
            }
#pragma unroll
            for (int j = 0; j < 4; ++j) {
                int it = tid + 256 * j;
                int row = it >> 3, c4 = (it & 7) * 4;
                unsigned short* d = lds_a + row * 32 + c4;
                d[0] = f2bf(f4[j].x); d[1] = f2bf(f4[j].y);
                d[2] = f2bf(f4[j].z); d[3] = f2bf(f4[j].w);
            }
        }
        // B tile: 128 rows (N=f) x 32 (K=e) bf16 — async copy straight to LDS
#pragma unroll
        for (int j = 0; j < 2; ++j) {
            int it = tid + 256 * j;                      // 512 b128-slots total
            int row = it >> 2, c8 = (it & 3) * 8;
            async_ld_b128(lds_b + row * 32 + c8,
                          wb + (size_t)(n0 + row) * EMB + ks + c8);
        }
        wait_async0();
        __syncthreads();

        v16bf af[2], bfr[4];
        af[0] = load_a_frag(lds_a + (wm +  0) * 32, 32, lane);
        af[1] = load_a_frag(lds_a + (wm + 16) * 32, 32, lane);
#pragma unroll
        for (int nt = 0; nt < 4; ++nt)
            bfr[nt] = load_b_frag_nk(lds_b + (wn + nt * 16) * 32, 32, lane);
#pragma unroll
        for (int mt = 0; mt < 2; ++mt)
#pragma unroll
            for (int nt = 0; nt < 4; ++nt)
                acc[mt][nt] = wmma_bf16(af[mt], bfr[nt], acc[mt][nt]);
    }

    // Epilogue: bias + scatter to [B,H,S,D] q/k/v
    const int h2 = lane >> 4, nlo = lane & 15;
#pragma unroll
    for (int mt = 0; mt < 2; ++mt)
#pragma unroll
        for (int nt = 0; nt < 4; ++nt) {
            int n = n0 + wn + nt * 16 + nlo;           // f in [0,3072)
            float bv = bias[n];
            int h = n / 192, rem = n - h * 192;
            int which = rem >> 6, d = rem & 63;
            unsigned short* dst = (which == 0) ? qb : ((which == 1) ? kb : vb);
#pragma unroll
            for (int v = 0; v < 8; ++v) {
                int m = r0 + wm + mt * 16 + v + 8 * h2;   // row = s*8+b
                int s = m >> 3, b = m & 7;
                dst[((size_t)(b * HEADS + h) * SEQ + s) * HDIM + d] =
                    f2bf(acc[mt][nt][v] + bv);
            }
        }
}

// ---------------------------------------------------------------------------
// Flash-style attention: one workgroup (4 waves) per (b*H+h, 64-row s block).
// ---------------------------------------------------------------------------
__global__ __launch_bounds__(128) void attn_kernel(const unsigned short* __restrict__ qb,
                                                   const unsigned short* __restrict__ kb,
                                                   const unsigned short* __restrict__ vb,
                                                   unsigned short* __restrict__ ctx) {
    __shared__ unsigned short lds_k[64 * 64];    // [t][d]
    __shared__ unsigned short lds_vT[64 * 64];   // [d][t]
    __shared__ unsigned short lds_p[4 * 16 * 64];// per-wave P staging [m][t]

    const int tid = threadIdx.x, lane = tid & 31, wave = tid >> 5;
    const int bh = blockIdx.y;
    const int s0 = blockIdx.x * 64 + wave * 16;
    const int h2 = lane >> 4, nlo = lane & 15;

    // Q fragments for this wave's 16 rows (K = d, two 32-chunks)
    const unsigned short* qrow = qb + ((size_t)bh * SEQ + s0) * HDIM;
    v16bf qf[2];
    qf[0] = load_a_frag(qrow,      HDIM, lane);
    qf[1] = load_a_frag(qrow + 32, HDIM, lane);

    const v8f vzero = {0.f, 0.f, 0.f, 0.f, 0.f, 0.f, 0.f, 0.f};
    v8f acc[4];
#pragma unroll
    for (int nt = 0; nt < 4; ++nt) acc[nt] = vzero;
    float mrun[8], lrun[8];
#pragma unroll
    for (int v = 0; v < 8; ++v) { mrun[v] = -3.0e30f; lrun[v] = 0.f; }

    for (int t0 = 0; t0 < SEQ; t0 += 64) {
        __syncthreads();
        // stage K tile [t][d] via async global->LDS DMA
        const unsigned short* ksrc = kb + ((size_t)bh * SEQ + t0) * HDIM;
#pragma unroll
        for (int j = 0; j < 4; ++j) {
            int it = tid + 128 * j;                 // 512 b128-slots
            int row = it >> 3, c8 = (it & 7) * 8;
            async_ld_b128(lds_k + row * 64 + c8, ksrc + row * 64 + c8);
        }
        // stage V tile transposed -> [d][t] (transpose needs the VGPR path)
        const unsigned short* vsrc = vb + ((size_t)bh * SEQ + t0) * HDIM;
#pragma unroll
        for (int j = 0; j < 16; ++j) {
            int it = tid + 128 * j;                 // 2048 uint-slots
            int t = it >> 5, d2 = (it & 31) * 2;
            unsigned int u = *(const unsigned int*)(vsrc + t * 64 + d2);
            lds_vT[d2 * 64 + t]       = (unsigned short)(u & 0xFFFFu);
            lds_vT[(d2 + 1) * 64 + t] = (unsigned short)(u >> 16);
        }
        wait_async0();
        __syncthreads();

        // scores: S = Q * K^T  (B operand rows = t, K-dim = d)
        v8f sc[4];
#pragma unroll
        for (int nt = 0; nt < 4; ++nt) {
            sc[nt] = vzero;
#pragma unroll
            for (int kk = 0; kk < 2; ++kk) {
                v16bf bfr = load_b_frag_nk(lds_k + (nt * 16) * 64 + kk * 32, 64, lane);
                sc[nt] = wmma_bf16(qf[kk], bfr, sc[nt]);
            }
        }

        // scale + online softmax (rows live in (lane/16, v); cols across 16 lanes)
        float tmax[8];
#pragma unroll
        for (int v = 0; v < 8; ++v) {
            float a = fmaxf(sc[0][v] , sc[1][v]);
            float b = fmaxf(sc[2][v] , sc[3][v]);
            tmax[v] = 0.125f * fmaxf(a, b);
        }
#pragma unroll
        for (int off = 1; off < 16; off <<= 1)
#pragma unroll
            for (int v = 0; v < 8; ++v)
                tmax[v] = fmaxf(tmax[v], __shfl_xor(tmax[v], off, 32));

#pragma unroll
        for (int v = 0; v < 8; ++v) {
            float mnew = fmaxf(mrun[v], tmax[v]);
            float corr = __expf(mrun[v] - mnew);
            float psum = 0.f;
#pragma unroll
            for (int nt = 0; nt < 4; ++nt) {
                float p = __expf(sc[nt][v] * 0.125f - mnew);
                sc[nt][v] = p;
                psum += p;
            }
#pragma unroll
            for (int off = 1; off < 16; off <<= 1)
                psum += __shfl_xor(psum, off, 32);
            lrun[v] = lrun[v] * corr + psum;
            mrun[v] = mnew;
#pragma unroll
            for (int nt = 0; nt < 4; ++nt) acc[nt][v] *= corr;
        }

        // P -> bf16 A-fragment via per-wave LDS patch
        unsigned short* pw = lds_p + wave * (16 * 64);
#pragma unroll
        for (int nt = 0; nt < 4; ++nt)
#pragma unroll
            for (int v = 0; v < 8; ++v)
                pw[(v + 8 * h2) * 64 + nt * 16 + nlo] = f2bf(sc[nt][v]);
        asm volatile("s_wait_dscnt 0" ::: "memory");   // wave-local LDS RAW

        v16bf pf[2];
        pf[0] = load_a_frag(pw,      64, lane);
        pf[1] = load_a_frag(pw + 32, 64, lane);

        // ctx += P * V  (B operand rows = d via lds_vT, K-dim = t)
#pragma unroll
        for (int nt = 0; nt < 4; ++nt)
#pragma unroll
            for (int kk = 0; kk < 2; ++kk) {
                v16bf bfr = load_b_frag_nk(lds_vT + (nt * 16) * 64 + kk * 32, 64, lane);
                acc[nt] = wmma_bf16(pf[kk], bfr, acc[nt]);
            }
    }

    // normalize and store ctx bf16 [B,H,S,D]
#pragma unroll
    for (int nt = 0; nt < 4; ++nt)
#pragma unroll
        for (int v = 0; v < 8; ++v) {
            float o = acc[nt][v] / lrun[v];
            ctx[((size_t)bh * SEQ + s0 + v + 8 * h2) * HDIM + nt * 16 + nlo] = f2bf(o);
        }
}

// ---------------------------------------------------------------------------
// Output projection + bias + residual: y = x + ctx * Wout^T + b  (fp32 out)
// ---------------------------------------------------------------------------
__global__ __launch_bounds__(256) void out_gemm_kernel(const unsigned short* __restrict__ ctx,
                                                       const unsigned short* __restrict__ wb,
                                                       const float* __restrict__ bias,
                                                       const float* __restrict__ x,
                                                       float* __restrict__ y) {
    __shared__ unsigned short lds_a[128 * 32];
    __shared__ unsigned short lds_b[128 * 32];

    const int tid = threadIdx.x, lane = tid & 31, wave = tid >> 5;
    const int n0 = blockIdx.x * 128;
    const int r0 = blockIdx.y * 128;
    const int wm = (wave >> 1) * 32, wn = (wave & 1) * 64;

    v8f acc[2][4];
    const v8f vzero = {0.f, 0.f, 0.f, 0.f, 0.f, 0.f, 0.f, 0.f};
#pragma unroll
    for (int mt = 0; mt < 2; ++mt)
#pragma unroll
        for (int nt = 0; nt < 4; ++nt) acc[mt][nt] = vzero;

    for (int ks = 0; ks < EMB; ks += 32) {
        __syncthreads();
        // A tile from ctx[B,H,S,D] (bf16, verbatim) — async copy to LDS
        const int h = ks >> 6, d0 = ks & 63;
#pragma unroll
        for (int j = 0; j < 2; ++j) {
            int it = tid + 256 * j;
            int row = it >> 2, c8 = (it & 3) * 8;
            int m = r0 + row, s = m >> 3, b = m & 7;
            async_ld_b128(lds_a + row * 32 + c8,
                          ctx + ((size_t)(b * HEADS + h) * SEQ + s) * HDIM + d0 + c8);
        }
        // B tile from out_w bf16 — async copy to LDS
#pragma unroll
        for (int j = 0; j < 2; ++j) {
            int it = tid + 256 * j;
            int row = it >> 2, c8 = (it & 3) * 8;
            async_ld_b128(lds_b + row * 32 + c8,
                          wb + (size_t)(n0 + row) * EMB + ks + c8);
        }
        wait_async0();
        __syncthreads();

        v16bf af[2], bfr[4];
        af[0] = load_a_frag(lds_a + (wm +  0) * 32, 32, lane);
        af[1] = load_a_frag(lds_a + (wm + 16) * 32, 32, lane);
#pragma unroll
        for (int nt = 0; nt < 4; ++nt)
            bfr[nt] = load_b_frag_nk(lds_b + (wn + nt * 16) * 32, 32, lane);
#pragma unroll
        for (int mt = 0; mt < 2; ++mt)
#pragma unroll
            for (int nt = 0; nt < 4; ++nt)
                acc[mt][nt] = wmma_bf16(af[mt], bfr[nt], acc[mt][nt]);
    }

    const int h2 = lane >> 4, nlo = lane & 15;
#pragma unroll
    for (int mt = 0; mt < 2; ++mt)
#pragma unroll
        for (int nt = 0; nt < 4; ++nt) {
            int n = n0 + wn + nt * 16 + nlo;
            float bv = bias[n];
#pragma unroll
            for (int v = 0; v < 8; ++v) {
                int m = r0 + wm + mt * 16 + v + 8 * h2;
                size_t idx = (size_t)m * EMB + n;
                y[idx] = acc[mt][nt][v] + bv + x[idx];
            }
        }
}

// ---------------------------------------------------------------------------
// LayerNorm + ELU over the last dim (1024), one block per row
// ---------------------------------------------------------------------------
__global__ __launch_bounds__(256) void ln_elu_kernel(const float* __restrict__ y,
                                                     const float* __restrict__ w,
                                                     const float* __restrict__ b,
                                                     float* __restrict__ out) {
    const int r = blockIdx.x, tid = threadIdx.x;
    const int lane = tid & 31, wave = tid >> 5;
    const float* row = y + (size_t)r * EMB;

    float vals[4];
    float s1 = 0.f, s2 = 0.f;
#pragma unroll
    for (int j = 0; j < 4; ++j) {
        float t = row[tid + 256 * j];
        vals[j] = t;
        s1 += t;
        s2 += t * t;
    }
#pragma unroll
    for (int off = 1; off < 32; off <<= 1) {
        s1 += __shfl_xor(s1, off, 32);
        s2 += __shfl_xor(s2, off, 32);
    }
    __shared__ float r1[8], r2[8];
    if (lane == 0) { r1[wave] = s1; r2[wave] = s2; }
    __syncthreads();
    float S1 = 0.f, S2 = 0.f;
#pragma unroll
    for (int i = 0; i < 8; ++i) { S1 += r1[i]; S2 += r2[i]; }

    const float mean = S1 * (1.0f / EMB);
    const float var  = S2 * (1.0f / EMB) - mean * mean;
    const float inv  = rsqrtf(var + 1e-5f);
#pragma unroll
    for (int j = 0; j < 4; ++j) {
        int c = tid + 256 * j;
        float t = (vals[j] - mean) * inv * w[c] + b[c];
        out[(size_t)r * EMB + c] = (t > 0.f) ? t : expm1f(t);
    }
}

// ---------------------------------------------------------------------------
// launch
// ---------------------------------------------------------------------------
extern "C" void kernel_launch(void* const* d_in, const int* in_sizes, int n_in,
                              void* d_out, int out_size, void* d_ws, size_t ws_size,
                              hipStream_t stream) {
    const float* x      = (const float*)d_in[0];
    const float* qkv_w  = (const float*)d_in[1];
    const float* qkv_b  = (const float*)d_in[2];
    const float* out_w  = (const float*)d_in[3];
    const float* out_b  = (const float*)d_in[4];
    const float* norm_w = (const float*)d_in[5];
    const float* norm_b = (const float*)d_in[6];
    float* out = (float*)d_out;

    char* ws = (char*)d_ws;
    // workspace layout (bytes):
    //   [0, 6291456)        qkv_w bf16  (3072*1024)
    //   [6291456, 8388608)  out_w bf16  (1024*1024)
    //   then q, k, v, ctx   bf16 [B,H,S,D] = 8388608 elems (16 MB) each
    //   y fp32 (33.5 MB) aliases q+k after attention is done
    unsigned short* qkvw_bf = (unsigned short*)ws;
    unsigned short* outw_bf = (unsigned short*)(ws + 6291456);
    unsigned short* qb      = (unsigned short*)(ws + 8388608);
    unsigned short* kb      = qb + 8388608;
    unsigned short* vb      = kb + 8388608;
    unsigned short* ctx     = vb + 8388608;
    float*          ybuf    = (float*)qb;

    cvt_bf16_kernel<<<(3 * EMB * EMB + 255) / 256, 256, 0, stream>>>(qkv_w, qkvw_bf, 3 * EMB * EMB);
    cvt_bf16_kernel<<<(EMB * EMB + 255) / 256, 256, 0, stream>>>(out_w, outw_bf, EMB * EMB);

    qkv_gemm_kernel<<<dim3(3 * EMB / 128, ROWS / 128), 256, 0, stream>>>(
        x, qkvw_bf, qkv_b, qb, kb, vb);

    attn_kernel<<<dim3(SEQ / 64, BATCH * HEADS), 128, 0, stream>>>(qb, kb, vb, ctx);

    out_gemm_kernel<<<dim3(EMB / 128, ROWS / 128), 256, 0, stream>>>(
        ctx, outw_bf, out_b, x, ybuf);

    ln_elu_kernel<<<ROWS, 256, 0, stream>>>(ybuf, norm_w, norm_b, out);
}
